// TransformerModule_10445360464105
// MI455X (gfx1250) — matrix operand order repaired and checked by reading, not compiled, not run on hardware
//
#include <hip/hip_runtime.h>
#include <stdint.h>

typedef unsigned short u16;
typedef __attribute__((ext_vector_type(16))) __bf16 v16bf;
typedef __attribute__((ext_vector_type(8)))  float  v8f;

union Frag16 { v16bf v; unsigned int u[8]; };

#define WB  4
#define WS_ 2048
#define WD  256
#define WH  8
#define WKD 256
#define WFF 512
#define WM  (WB*WS_)    // 8192 rows
#define WHK (WH*WKD)    // 2048
#define LPAD 40         // padded LDS row stride in u16 (20 dwords: conflict-free, 16B aligned)
#define KPAD 264        // padded K-tile row stride in u16 (132 dwords: 16 distinct banks, 16B aligned)

// ---------- helpers ----------
static __device__ __forceinline__ u16 f2bf(float f) {
  unsigned int u = __float_as_uint(f);
  unsigned int r = u + 0x7FFFu + ((u >> 16) & 1u);
  return (u16)(r >> 16);
}
// 16-bit A-matrix 16x32: lanes0-15 row M=lane, K pairs: j0-3 -> 2j, j4-7 -> 16+2(j-4); lanes16-31 shift K by +8
static __device__ __forceinline__ int kbaseA(int j, int hi) {
  return 2*(j & 3) + ((j >= 4) ? 16 : 0) + (hi ? 8 : 0);
}
// B-matrix 32x16: lane = N, K pairs same as A: j0-3 -> 2j, j4-7 -> 16+2(j-4); lanes16-31 shift K by +8
static __device__ __forceinline__ int kbaseB(int j, int hi) {
  return 2*(j & 3) + ((j >= 4) ? 16 : 0) + (hi ? 8 : 0);
}
static __device__ __forceinline__ v8f vzero() {
  v8f z = {0.f,0.f,0.f,0.f,0.f,0.f,0.f,0.f};
  return z;
}
static __device__ __forceinline__ v8f wmma_bf16(const Frag16& a, const Frag16& b, v8f c) {
  return __builtin_amdgcn_wmma_f32_16x16x32_bf16(false, a.v, false, b.v, (short)0, c, false, false);
}
// CDNA5 async copy global -> LDS (16B per lane), tracked by ASYNCcnt
static __device__ __forceinline__ void async_ld_b128(unsigned lds_off, const u16* gptr) {
  asm volatile("global_load_async_to_lds_b128 %0, %1, off"
               :: "v"(lds_off), "v"(gptr) : "memory");
}
static __device__ __forceinline__ void wait_async0() {
  asm volatile("s_wait_asynccnt 0x0" ::: "memory");
}
static __device__ __forceinline__ void wait_async8() {
  asm volatile("s_wait_asynccnt 0x8" ::: "memory");
}
static __device__ __forceinline__ unsigned lds_addr_of(const void* p) {
  return (unsigned)(size_t)p;   // low 32 bits of a shared-aperture flat address = LDS offset
}

// ---------- conversion / transpose ----------
__global__ void k_f32_to_bf16(const float* __restrict__ src, u16* __restrict__ dst, int n) {
  int i = blockIdx.x * blockDim.x + threadIdx.x;
  if (i < n) dst[i] = f2bf(src[i]);
}

// src: f32 [K][N] row-major  ->  dst: bf16 [N][K]
__global__ void k_transpose_bf16(const float* __restrict__ src, u16* __restrict__ dst, int K, int N) {
  int i = blockIdx.x * blockDim.x + threadIdx.x;
  if (i < K * N) {
    int k = i / N, n = i - k * N;
    dst[(size_t)n * K + k] = f2bf(src[i]);
  }
}

// ---------- tiled bf16 WMMA GEMM:  C[M,N] = A[M,K] * BT[N,K]^T + bias[n]  ----------
// Double-buffered tiles staged with GLOBAL_LOAD_ASYNC_TO_LDS_B128 (ASYNCcnt),
// one barrier per K-step.
template<int MODE>
__global__ __launch_bounds__(256) void k_gemm(const u16* __restrict__ A, const u16* __restrict__ BT,
                       const float* __restrict__ bias, void* __restrict__ outp,
                       int M, int N, int K) {
  __shared__ u16 lA[2][64 * LPAD];
  __shared__ u16 lB[2][128 * LPAD];
  const int tid  = threadIdx.x;
  const int lane = tid & 31, wave = tid >> 5;
  const int hi   = lane >> 4, l16 = lane & 15;
  const int wm   = wave >> 2, wn = wave & 3;          // 2x4 wave grid
  const int mblk = blockIdx.y * 64, nblk = blockIdx.x * 128;
  const int m0   = mblk + wm * 32, n0 = nblk + wn * 32;

  v8f acc[2][2] = { { vzero(), vzero() }, { vzero(), vzero() } };

  const int arow = tid >> 2, ac = (tid & 3) * 8;      // 64 rows x 32 cols, 16B per thread
  const int brow = tid >> 1, bc = (tid & 1) * 16;     // 128 rows x 32 cols, 32B per thread
  const u16* agp = A  + (size_t)(mblk + arow) * K + ac;
  const u16* bgp = BT + (size_t)(nblk + brow) * K + bc;

  const unsigned lAoff0 = lds_addr_of(&lA[0][arow * LPAD + ac]);
  const unsigned lAoff1 = lds_addr_of(&lA[1][arow * LPAD + ac]);
  const unsigned lBoff0 = lds_addr_of(&lB[0][brow * LPAD + bc]);
  const unsigned lBoff1 = lds_addr_of(&lB[1][brow * LPAD + bc]);

  // prologue: async-copy tile 0 into buffer 0
  async_ld_b128(lAoff0, agp);
  async_ld_b128(lBoff0, bgp);
  async_ld_b128(lBoff0 + 16, bgp + 8);

  int buf = 0;
  for (int k0 = 0; k0 < K; k0 += 32) {
    wait_async0();
    __syncthreads();
    if (k0 + 32 < K) {        // kick off next tile into the other buffer
      const int kn = k0 + 32;
      async_ld_b128(buf ? lAoff0 : lAoff1, agp + kn);
      async_ld_b128((buf ? lBoff0 : lBoff1),      bgp + kn);
      async_ld_b128((buf ? lBoff0 : lBoff1) + 16, bgp + kn + 8);
    }
    const u16* cA = lA[buf];
    const u16* cB = lB[buf];
    Frag16 af[2], bfg[2];
#pragma unroll
    for (int j = 0; j < 8; ++j) {
      const int ka = kbaseA(j, hi), kb = kbaseB(j, hi);
      af[0].u[j]  = *(const unsigned int*)&cA[(wm * 32 +      l16) * LPAD + ka];
      af[1].u[j]  = *(const unsigned int*)&cA[(wm * 32 + 16 + l16) * LPAD + ka];
      bfg[0].u[j] = *(const unsigned int*)&cB[(wn * 32 +      l16) * LPAD + kb];
      bfg[1].u[j] = *(const unsigned int*)&cB[(wn * 32 + 16 + l16) * LPAD + kb];
    }
#pragma unroll
    for (int im = 0; im < 2; ++im)
#pragma unroll
      for (int in = 0; in < 2; ++in)
        acc[im][in] = wmma_bf16(af[im], bfg[in], acc[im][in]);
    buf ^= 1;
  }

  // epilogue
#pragma unroll
  for (int im = 0; im < 2; ++im) {
#pragma unroll
    for (int in = 0; in < 2; ++in) {
      const int n = n0 + in * 16 + l16;
      const float bv = bias[n];
#pragma unroll
      for (int r = 0; r < 8; ++r) {
        const int m = m0 + im * 16 + r + 8 * hi;     // C layout: M = r + 8*laneHi
        float val = acc[im][in][r] + bv;
        if (MODE == 0) {
          const int b = m >> 11, s = m & (WS_ - 1);
          const int h = n >> 8,  e = n & (WKD - 1);
          ((u16*)outp)[((size_t)(b * WH + h) * WS_ + s) * WKD + e] = f2bf(val * 0.0625f);
        } else if (MODE == 1) {
          const int b = m >> 11, s = m & (WS_ - 1);
          const int h = n >> 8,  e = n & (WKD - 1);
          ((u16*)outp)[((size_t)(b * WH + h) * WS_ + s) * WKD + e] = f2bf(val);
        } else if (MODE == 2) {
          const int b = m >> 11, s = m & (WS_ - 1);
          const int h = n >> 8,  e = n & (WKD - 1);
          ((u16*)outp)[((size_t)(b * WH + h) * WKD + e) * WS_ + s] = f2bf(val);
        } else if (MODE == 3 || MODE == 5) {
          ((float*)outp)[(size_t)m * N + n] = val;
        } else { // MODE 4: ReLU, bf16
          ((u16*)outp)[(size_t)m * N + n] = f2bf(fmaxf(val, 0.f));
        }
      }
    }
  }
}

// ---------- flash attention ----------
// All 4 waves of a block share (b,h) and the key loop, so K/V tiles are staged
// into LDS ONCE per block via async copies: K double-buffered (prefetched one
// iteration ahead), V single-buffered (copy overlaps the score phase; awaited
// with a partial asynccnt wait so the next K tile stays in flight).
// Wave pair per 16-query tile; each wave owns half of KD (8 O-fragments).
__global__ __launch_bounds__(128) void k_attention(const u16* __restrict__ qh, const u16* __restrict__ kh,
                            const u16* __restrict__ vT, u16* __restrict__ attnb) {
  __shared__ u16 ldsK[2][32 * KPAD];                  // 32 keys x 256 (padded)
  __shared__ u16 ldsV[256 * LPAD];                    // 256 e x 32 keys (padded)
  __shared__ u16 ldsp[4][16 * LPAD];                  // private P-transpose slice per wave
  const int tid   = threadIdx.x;
  const int lane  = tid & 31, wave = tid >> 5;
  const int hi    = lane >> 4, l16 = lane & 15;
  const int bh    = blockIdx.y;
  const int b     = bh >> 3, h = bh & 7;
  const int tile  = blockIdx.x * 2 + (wave >> 1);     // 2 query tiles per block
  const int q0    = tile * 16;
  const int eoff  = (wave & 1) * 128;                 // this wave's KD half

  const u16* qbase = qh + (size_t)bh * WS_ * WKD;
  const u16* kbase = kh + (size_t)bh * WS_ * WKD;
  const u16* vbase = vT + (size_t)bh * WKD * WS_;

  const unsigned adrK0 = lds_addr_of(&ldsK[0][0]);
  const unsigned adrK1 = lds_addr_of(&ldsK[1][0]);
  const unsigned adrV  = lds_addr_of(&ldsV[0]);

  // stage a 32x256 K tile (16 KB): 8 x 16B per thread
  auto issueK = [&](int key0, unsigned base) {
#pragma unroll
    for (int i = 0; i < 8; ++i) {
      const int linear = i * 128 + tid;
      const int row = linear >> 5, chunk = linear & 31;   // 32 chunks of 16B per row
      async_ld_b128(base + (unsigned)(row * KPAD + chunk * 8) * 2,
                    kbase + (size_t)(key0 + row) * WKD + chunk * 8);
    }
  };
  // stage a 256x32 V tile (16 KB): 8 x 16B per thread
  auto issueV = [&](int key0) {
#pragma unroll
    for (int i = 0; i < 8; ++i) {
      const int linear = i * 128 + tid;
      const int row = linear >> 2, chunk = linear & 3;    // 4 chunks of 16B per row
      async_ld_b128(adrV + (unsigned)(row * LPAD + chunk * 8) * 2,
                    vbase + (size_t)row * WS_ + key0 + chunk * 8);
    }
  };

  // resident Q A-fragments (16 queries x 256 = 8 fragments, 64 VGPRs)
  Frag16 qf[8];
  {
    const u16* qrow = qbase + (size_t)(q0 + l16) * WKD;
#pragma unroll
    for (int kc = 0; kc < 8; ++kc)
#pragma unroll
      for (int j = 0; j < 8; ++j)
        qf[kc].u[j] = *(const unsigned int*)&qrow[kc * 32 + kbaseA(j, hi)];
  }

  v8f o[8];
#pragma unroll
  for (int c = 0; c < 8; ++c) o[c] = vzero();
  float mrow[8], lrow[8];
#pragma unroll
  for (int r = 0; r < 8; ++r) { mrow[r] = -1e30f; lrow[r] = 0.f; }

  u16* lp = ldsp[wave];

  issueK(0, adrK0);                                   // prologue: K tile 0
  int buf = 0;
  for (int key0 = 0; key0 < WS_; key0 += 32) {
    const bool more = (key0 + 32 < WS_);
    wait_async0();                                    // this iter's K tile landed
    __syncthreads();
    issueV(key0);                                     // V first (awaited at cnt<=8)
    if (more) issueK(key0 + 32, buf ? adrK0 : adrK1); // K-next last (stays in flight)

    // scores for 32 keys from LDS: two 16x16 C-tiles (both waves of the pair)
    const u16* kt = ldsK[buf];
    v8f s0 = vzero(), s1 = vzero();
#pragma unroll
    for (int kc = 0; kc < 8; ++kc) {
      Frag16 kb0, kb1;
#pragma unroll
      for (int j = 0; j < 8; ++j) {
        const int kb = kc * 32 + kbaseB(j, hi);
        kb0.u[j] = *(const unsigned int*)&kt[       l16  * KPAD + kb];
        kb1.u[j] = *(const unsigned int*)&kt[(16 + l16) * KPAD + kb];
      }
      s0 = wmma_bf16(qf[kc], kb0, s0);
      s1 = wmma_bf16(qf[kc], kb1, s1);
    }

    // online softmax (row = r + 8*hi, cols across the 16-lane half)
    float alpha[8];
#pragma unroll
    for (int r = 0; r < 8; ++r) {
      float t = fmaxf(s0[r], s1[r]);
      t = fmaxf(t, __shfl_xor(t, 1, 32));
      t = fmaxf(t, __shfl_xor(t, 2, 32));
      t = fmaxf(t, __shfl_xor(t, 4, 32));
      t = fmaxf(t, __shfl_xor(t, 8, 32));
      const float mn = fmaxf(mrow[r], t);
      alpha[r] = __expf(mrow[r] - mn);
      mrow[r]  = mn;
      const float p0 = __expf(s0[r] - mn);
      const float p1 = __expf(s1[r] - mn);
      float rs = p0 + p1;
      rs += __shfl_xor(rs, 1, 32);
      rs += __shfl_xor(rs, 2, 32);
      rs += __shfl_xor(rs, 4, 32);
      rs += __shfl_xor(rs, 8, 32);
      lrow[r] = lrow[r] * alpha[r] + rs;
      const int row = r + 8 * hi;
      lp[row * LPAD +      l16] = f2bf(p0);
      lp[row * LPAD + 16 + l16] = f2bf(p1);
    }
#pragma unroll
    for (int c = 0; c < 8; ++c)
#pragma unroll
      for (int r = 0; r < 8; ++r)
        o[c][r] *= alpha[r];

    if (more) wait_async8(); else wait_async0();      // V landed (K-next may fly)
    __syncthreads();

    Frag16 pa;
#pragma unroll
    for (int j = 0; j < 8; ++j)
      pa.u[j] = *(const unsigned int*)&lp[l16 * LPAD + kbaseA(j, hi)];

    // O += P * V on this wave's KD half, V from LDS (K = key pairs contiguous)
#pragma unroll
    for (int c = 0; c < 8; ++c) {
      Frag16 vb;
      const u16* vrow = &ldsV[(eoff + c * 16 + l16) * LPAD];
#pragma unroll
      for (int j = 0; j < 8; ++j)
        vb.u[j] = *(const unsigned int*)&vrow[kbaseB(j, hi)];
      o[c] = wmma_bf16(pa, vb, o[c]);
    }
    buf ^= 1;
  }

  // normalize and scatter to attn[b,s,h,e] (bf16, row-major [m][H*KD])
  float invl[8];
#pragma unroll
  for (int r = 0; r < 8; ++r) invl[r] = 1.f / lrow[r];
#pragma unroll
  for (int c = 0; c < 8; ++c) {
    const int e = eoff + c * 16 + l16;
#pragma unroll
    for (int r = 0; r < 8; ++r) {
      const int m = b * WS_ + q0 + r + 8 * hi;
      attnb[(size_t)m * WHK + h * WKD + e] = f2bf(o[c][r] * invl[r]);
    }
  }
}

// ---------- LayerNorm kernels (1 wave per row of 256) ----------
__global__ void k_ln1(const float* __restrict__ lin, const float* __restrict__ x,
                      const float* __restrict__ g, const float* __restrict__ bt,
                      float* __restrict__ z, u16* __restrict__ zb) {
  const int row = blockIdx.x, lane = threadIdx.x;
  const float* p = lin + (size_t)row * WD;
  float v[8], s = 0.f, ss = 0.f;
#pragma unroll
  for (int i = 0; i < 8; ++i) { v[i] = p[lane + i * 32]; s += v[i]; ss += v[i] * v[i]; }
#pragma unroll
  for (int off = 16; off >= 1; off >>= 1) { s += __shfl_xor(s, off, 32); ss += __shfl_xor(ss, off, 32); }
  const float mean = s * (1.f / WD);
  const float var  = fmaxf(ss * (1.f / WD) - mean * mean, 0.f);
  const float rstd = rsqrtf(var + 1e-3f);
#pragma unroll
  for (int i = 0; i < 8; ++i) {
    const int n = lane + i * 32;
    const float t = (v[i] - mean) * rstd * g[n] + bt[n] + x[(size_t)row * WD + n];
    z[(size_t)row * WD + n]  = t;
    zb[(size_t)row * WD + n] = f2bf(t);
  }
}

__global__ void k_ln2(const float* __restrict__ f, const float* __restrict__ z,
                      const float* __restrict__ g, const float* __restrict__ bt,
                      float* __restrict__ out) {
  const int row = blockIdx.x, lane = threadIdx.x;
  float v[8], s = 0.f, ss = 0.f;
#pragma unroll
  for (int i = 0; i < 8; ++i) {
    const int n = lane + i * 32;
    v[i] = f[(size_t)row * WD + n] + z[(size_t)row * WD + n];
    s += v[i]; ss += v[i] * v[i];
  }
#pragma unroll
  for (int off = 16; off >= 1; off >>= 1) { s += __shfl_xor(s, off, 32); ss += __shfl_xor(ss, off, 32); }
  const float mean = s * (1.f / WD);
  const float var  = fmaxf(ss * (1.f / WD) - mean * mean, 0.f);
  const float rstd = rsqrtf(var + 1e-3f);
#pragma unroll
  for (int i = 0; i < 8; ++i) {
    const int n = lane + i * 32;
    out[(size_t)row * WD + n] = (v[i] - mean) * rstd * g[n] + bt[n];
  }
}

// ---------- host ----------
extern "C" void kernel_launch(void* const* d_in, const int* in_sizes, int n_in,
                              void* d_out, int out_size, void* d_ws, size_t ws_size,
                              hipStream_t stream) {
  (void)in_sizes; (void)n_in; (void)out_size; (void)ws_size;
  const float* x   = (const float*)d_in[0];
  const float* Wq  = (const float*)d_in[1];
  const float* bq  = (const float*)d_in[2];
  const float* Wk  = (const float*)d_in[3];
  const float* bk  = (const float*)d_in[4];
  const float* Wv  = (const float*)d_in[5];
  const float* bv  = (const float*)d_in[6];
  const float* Wo  = (const float*)d_in[7];
  const float* bo  = (const float*)d_in[8];
  const float* g1  = (const float*)d_in[9];
  const float* be1 = (const float*)d_in[10];
  const float* W1  = (const float*)d_in[11];
  const float* b1  = (const float*)d_in[12];
  const float* W2  = (const float*)d_in[13];
  const float* b2  = (const float*)d_in[14];
  const float* g2  = (const float*)d_in[15];
  const float* be2 = (const float*)d_in[16];

  char* ws = (char*)d_ws;
  size_t off = 0;
  auto alloc = [&](size_t bytes) -> char* {
    char* p = ws + off;
    off += (bytes + 255) & ~(size_t)255;
    return p;
  };
  u16*   xb    = (u16*)  alloc((size_t)WM * WD * 2);
  u16*   WqT   = (u16*)  alloc((size_t)WHK * WD * 2);
  u16*   WkT   = (u16*)  alloc((size_t)WHK * WD * 2);
  u16*   WvT   = (u16*)  alloc((size_t)WHK * WD * 2);
  u16*   WoT   = (u16*)  alloc((size_t)WD * WHK * 2);
  u16*   W1T   = (u16*)  alloc((size_t)WFF * WD * 2);
  u16*   W2T   = (u16*)  alloc((size_t)WD * WFF * 2);
  u16*   qh    = (u16*)  alloc((size_t)WB * WH * WS_ * WKD * 2);
  u16*   khb   = (u16*)  alloc((size_t)WB * WH * WS_ * WKD * 2);
  u16*   vTb   = (u16*)  alloc((size_t)WB * WH * WKD * WS_ * 2);
  u16*   attnb = (u16*)  alloc((size_t)WM * WHK * 2);
  float* alin  = (float*)alloc((size_t)WM * WD * 4);
  float* zbuf  = (float*)alloc((size_t)WM * WD * 4);
  u16*   zb    = (u16*)  alloc((size_t)WM * WD * 2);
  u16*   hbuf  = (u16*)  alloc((size_t)WM * WFF * 2);
  float* fbuf  = (float*)alloc((size_t)WM * WD * 4);

  // stage bf16 operands
  k_f32_to_bf16<<<(WM * WD) / 256, 256, 0, stream>>>(x, xb, WM * WD);
  k_transpose_bf16<<<(WD * WHK) / 256, 256, 0, stream>>>(Wq, WqT, WD, WHK);
  k_transpose_bf16<<<(WD * WHK) / 256, 256, 0, stream>>>(Wk, WkT, WD, WHK);
  k_transpose_bf16<<<(WD * WHK) / 256, 256, 0, stream>>>(Wv, WvT, WD, WHK);
  k_transpose_bf16<<<(WHK * WD) / 256, 256, 0, stream>>>(Wo, WoT, WHK, WD);
  k_transpose_bf16<<<(WD * WFF) / 256, 256, 0, stream>>>(W1, W1T, WD, WFF);
  k_transpose_bf16<<<(WFF * WD) / 256, 256, 0, stream>>>(W2, W2T, WFF, WD);

  // QKV projections
  dim3 gqkv(WHK / 128, WM / 64);
  k_gemm<0><<<gqkv, 256, 0, stream>>>(xb, WqT, bq, qh,  WM, WHK, WD);
  k_gemm<1><<<gqkv, 256, 0, stream>>>(xb, WkT, bk, khb, WM, WHK, WD);
  k_gemm<2><<<gqkv, 256, 0, stream>>>(xb, WvT, bv, vTb, WM, WHK, WD);

  // flash attention (2 query tiles per 128-thread block, wave pairs split KD)
  k_attention<<<dim3(WS_ / 32, WB * WH), 128, 0, stream>>>(qh, khb, vTb, attnb);

  // output projection + LN1 + residual
  k_gemm<3><<<dim3(WD / 128, WM / 64), 256, 0, stream>>>(attnb, WoT, bo, alin, WM, WD, WHK);
  k_ln1<<<WM, 32, 0, stream>>>(alin, x, g1, be1, zbuf, zb);

  // FFN + LN2
  k_gemm<4><<<dim3(WFF / 128, WM / 64), 256, 0, stream>>>(zb, W1T, b1, hbuf, WM, WFF, WD);
  k_gemm<5><<<dim3(WD / 128, WM / 64), 256, 0, stream>>>(hbuf, W2T, b2, fbuf, WM, WD, WFF);
  k_ln2<<<WM, 32, 0, stream>>>(fbuf, zbuf, g2, be2, (float*)d_out);
}
